// TreeLSTMCell_53506702573723
// MI455X (gfx1250) — matrix operand order repaired and checked by reading, not compiled, not run on hardware
//
#include <hip/hip_runtime.h>
#include <hip/hip_bf16.h>

// ---------------------------------------------------------------------------
// TreeLSTM cell, fused as one bf16 WMMA GEMM:
//   A = [x | child_h[child_idx]]            [N, 512]   (bf16, staged in LDS)
//   B = [[W_iou|U_iou],[W_f|U_f]]           [1024, 512] (bf16, K-major in ws)
//   G = A @ B^T  -> i,o,u columns 0..767, f columns 768..1023
// Epilogue (wave-local): c = sig(i)*tanh(u) + sig(f)*child_c[idx]; h = sig(o)*tanh(c)
// ---------------------------------------------------------------------------

#define N_NODES 131072
#define BM      64          // rows per workgroup
#define LDA     520         // padded LDS row stride (bf16 elems): 520*2B = 65*16B, bank-skew 4

typedef __attribute__((ext_vector_type(16))) __bf16 v16bf;
typedef __attribute__((ext_vector_type(8)))  float  v8f;
typedef unsigned int u32x4 __attribute__((ext_vector_type(4)));
typedef unsigned int u32x2 __attribute__((ext_vector_type(2)));
typedef float        f32x4 __attribute__((ext_vector_type(4)));

union FragU { u32x4 q[2]; v16bf v; };

static __device__ __forceinline__ unsigned int f2bf(float f) {
  union { float f; unsigned int u; } c; c.f = f;
  unsigned int u = c.u;
  u += 0x7FFFu + ((u >> 16) & 1u);          // round-to-nearest-even
  return u >> 16;
}

static __device__ __forceinline__ float sigf(float x) {
  return 1.0f / (1.0f + __expf(-x));
}

// ---- Phase 0: convert & repack weights to bf16, K-major: Bw[k][n], k<512, n<1024
__global__ void convert_weights_kernel(const float* __restrict__ Wiou,
                                       const float* __restrict__ Uiou,
                                       const float* __restrict__ Wf,
                                       const float* __restrict__ Uf,
                                       unsigned short* __restrict__ Bw) {
  int idx = blockIdx.x * blockDim.x + threadIdx.x;   // 0 .. 512*1024-1
  int k = idx >> 10;
  int n = idx & 1023;
  float v;
  if (n < 768) {
    v = (k < 256) ? Wiou[n * 256 + k] : Uiou[n * 256 + (k - 256)];
  } else {
    int nf = n - 768;
    v = (k < 256) ? Wf[nf * 256 + k] : Uf[nf * 256 + (k - 256)];
  }
  Bw[idx] = (unsigned short)f2bf(v);
}

// ---- Phase 1: fused GEMM + gates. 512 threads = 16 waves.
// wave = (rowhalf<<3) | colgroup; each wave: 32 rows x 32 gate-relative cols,
// all 4 gate blocks -> 2(mt) x 4(gate) x 2(nt) = 16 WMMA accum tiles.
__global__ __launch_bounds__(512)
void treelstm_kernel(const float* __restrict__ x,
                     const float* __restrict__ child_h,
                     const float* __restrict__ child_c,
                     const int*   __restrict__ child_idx,
                     const float* __restrict__ b_iou,
                     const float* __restrict__ W_f_b,
                     const float* __restrict__ U_f_b,
                     const float* __restrict__ b_f,
                     const unsigned short* __restrict__ Bw,
                     float* __restrict__ out) {
  __shared__ __align__(16) unsigned short As[BM * LDA];
  __shared__ int sidx[BM];

  const int tid = threadIdx.x;
  const int m0  = blockIdx.x * BM;

  // ---------------- stage A panel (x | gathered child_h) as bf16 ----------------
  {
    const int r   = tid >> 3;          // 0..63 local row
    const int s   = tid & 7;           // 8 threads per row
    const int row = m0 + r;
    const int gi  = child_idx[row];
    if (s == 0) sidx[r] = gi;
    const f32x4* xr = reinterpret_cast<const f32x4*>(x + (size_t)row * 256);
    const f32x4* hr = reinterpret_cast<const f32x4*>(child_h + (size_t)gi * 256);
    u32x2* dst2 = reinterpret_cast<u32x2*>(&As[r * LDA]);
#pragma unroll
    for (int i = 0; i < 8; ++i) {
      f32x4 f = xr[s + i * 8];
      int e = (s + i * 8) * 4;                         // col 0..255 (x part)
      u32x2 p = { f2bf(f.x) | (f2bf(f.y) << 16),
                  f2bf(f.z) | (f2bf(f.w) << 16) };
      dst2[e >> 2] = p;
    }
#pragma unroll
    for (int i = 0; i < 8; ++i) {
      f32x4 f = hr[s + i * 8];
      int e = 256 + (s + i * 8) * 4;                   // col 256..511 (hc part)
      u32x2 p = { f2bf(f.x) | (f2bf(f.y) << 16),
                  f2bf(f.z) | (f2bf(f.w) << 16) };
      dst2[e >> 2] = p;
    }
  }
  __syncthreads();

  // ---------------- K loop: 16 WMMA per step ----------------
  const int lane = tid & 31;
  const int wv   = tid >> 5;          // 0..15
  const int cg   = wv & 7;            // column group
  const int rh   = wv >> 3;           // row half
  const int j0   = cg * 32;           // gate-relative column base

  v8f acc[2][4][2];
  {
    v8f z = {0.f, 0.f, 0.f, 0.f, 0.f, 0.f, 0.f, 0.f};
#pragma unroll
    for (int mt = 0; mt < 2; ++mt)
#pragma unroll
      for (int g = 0; g < 4; ++g)
#pragma unroll
        for (int nt = 0; nt < 2; ++nt) acc[mt][g][nt] = z;
  }

  const int koff  = (lane >> 4) << 3;             // hi lanes: K offset +8
  const int arow0 = (lane & 15) + rh * 32;        // A row for this lane (mt adds 16)

  for (int k0 = 0; k0 < 512; k0 += 32) {
    FragU fa[2];
#pragma unroll
    for (int mt = 0; mt < 2; ++mt) {
      const u32x4* ap = reinterpret_cast<const u32x4*>(
          &As[(arow0 + mt * 16) * LDA + k0 + koff]);
      fa[mt].q[0] = ap[0];        // K {k0..k0+7}   (+8 for hi lanes)
      fa[mt].q[1] = ap[2];        // K {k0+16..k0+23}
    }
    const unsigned short* brow = Bw + (size_t)(k0 + koff) * 1024 + (lane & 15);  // lane = N index
#pragma unroll
    for (int g = 0; g < 4; ++g) {
#pragma unroll
      for (int nt = 0; nt < 2; ++nt) {
        const unsigned short* bp = brow + g * 256 + j0 + nt * 16;
        FragU fb;
#pragma unroll
        for (int e = 0; e < 4; ++e) {
          fb.q[0][e] = (unsigned int)bp[(2 * e) * 1024] | ((unsigned int)bp[(2 * e + 1) * 1024] << 16);
          fb.q[1][e] = (unsigned int)bp[(16 + 2 * e) * 1024] | ((unsigned int)bp[(17 + 2 * e) * 1024] << 16);
        }
#pragma unroll
        for (int mt = 0; mt < 2; ++mt) {
          acc[mt][g][nt] = __builtin_amdgcn_wmma_f32_16x16x32_bf16(
              false, fa[mt].v, false, fb.v, (short)0, acc[mt][g][nt],
              false, false);
        }
      }
    }
  }

  // ---------------- wave-local gate epilogue ----------------
  float* out_h = out;
  float* out_c = out + (size_t)N_NODES * 256;

#pragma unroll
  for (int nt = 0; nt < 2; ++nt) {
    const int col = j0 + nt * 16 + (lane & 15);      // 0..255 gate-relative
    const float bi  = b_iou[col];
    const float bo  = b_iou[256 + col];
    const float bu  = b_iou[512 + col];
    const float bfv = W_f_b[col] + U_f_b[col] + b_f[col];
#pragma unroll
    for (int mt = 0; mt < 2; ++mt) {
      const int rloc0 = rh * 32 + mt * 16 + ((lane >> 4) << 3);
#pragma unroll
      for (int v = 0; v < 8; ++v) {
        const int rloc = rloc0 + v;
        const int row  = m0 + rloc;
        const float iv = acc[mt][0][nt][v] + bi;
        const float ov = acc[mt][1][nt][v] + bo;
        const float uv = acc[mt][2][nt][v] + bu;
        const float fv = acc[mt][3][nt][v] + bfv;
        const int   gi = sidx[rloc];
        const float cc = child_c[(size_t)gi * 256 + col];
        const float cv = sigf(iv) * tanhf(uv) + sigf(fv) * cc;
        const float hv = sigf(ov) * tanhf(cv);
        out_h[(size_t)row * 256 + col] = hv;
        out_c[(size_t)row * 256 + col] = cv;
      }
    }
  }
}

extern "C" void kernel_launch(void* const* d_in, const int* in_sizes, int n_in,
                              void* d_out, int out_size, void* d_ws, size_t ws_size,
                              hipStream_t stream) {
  const float* x         = (const float*)d_in[0];
  const float* child_h   = (const float*)d_in[1];
  const float* child_c   = (const float*)d_in[2];
  const int*   child_idx = (const int*)d_in[3];
  const float* W_iou_w   = (const float*)d_in[4];
  const float* U_iou_w   = (const float*)d_in[5];
  const float* b_iou     = (const float*)d_in[6];
  const float* W_f_w     = (const float*)d_in[7];
  const float* W_f_b     = (const float*)d_in[8];
  const float* U_f_w     = (const float*)d_in[9];
  const float* U_f_b     = (const float*)d_in[10];
  const float* b_f       = (const float*)d_in[11];

  // ws layout: Bw[512][1024] bf16 = 1 MiB (K-major combined weight matrix)
  unsigned short* Bw = (unsigned short*)d_ws;

  convert_weights_kernel<<<(512 * 1024) / 256, 256, 0, stream>>>(
      W_iou_w, U_iou_w, W_f_w, U_f_w, Bw);

  treelstm_kernel<<<N_NODES / BM, 512, 0, stream>>>(
      x, child_h, child_c, child_idx, b_iou, W_f_b, U_f_b, b_f, Bw,
      (float*)d_out);
}
